// DCP_matching_one2one_76544907149617
// MI455X (gfx1250) — compile-verified, matrix operand order checked by reading, not run on hardware
//
#include <hip/hip_runtime.h>
#include <hip/hip_bf16.h>
#include <math.h>

typedef __attribute__((ext_vector_type(16))) _Float16 v16h;
typedef __attribute__((ext_vector_type(8)))  float    v8f;

#define BB 16
#define DD 512
#define NN 2048
#define SS 15
#define KC (DD / 32)   // 16 k-chunks of 32
#define NT (NN / 16)   // 128 tiles of 16

// ---------------------------------------------------------------------------
// Pack f32 [D][N] embedding into f16 WMMA fragment-native layout.
// Fragment f = ((b*KC + kc)*NT + nt); per fragment: 32 lanes x 16 halves.
// A/B 16-bit 16x32 layout: lane L holds M = L&15;
//   K(L,h) = (h<8 ? 0 : 16) + (L>=16 ? 8 : 0) + (h&7)   (within the 32-chunk)
// ---------------------------------------------------------------------------
__global__ void pack_frags_kernel(const float* __restrict__ emb,
                                  _Float16* __restrict__ pack) {
  int f = blockIdx.x * 8 + (threadIdx.x >> 5);
  int lane = threadIdx.x & 31;
  if (f >= BB * KC * NT) return;
  int nt = f % NT;
  int kc = (f / NT) % KC;
  int b  = f / (NT * KC);
  int m = nt * 16 + (lane & 15);
  const float* src = emb + (size_t)b * DD * NN;
  _Float16* dst = pack + ((size_t)f * 32 + lane) * 16;
  int kb = (lane >= 16) ? 8 : 0;
#pragma unroll
  for (int h = 0; h < 16; ++h) {
    int k = kc * 32 + ((h < 8) ? 0 : 16) + kb + (h & 7);
    dst[h] = (_Float16)src[(size_t)k * NN + m];
  }
}

// ---------------------------------------------------------------------------
// GEMM: logits[b][s][t] = scale * sum_d A[b][d][s]*B[b][d][t]
// 256 threads = 8 waves; each wave owns one sTile x 4 consecutive tTiles:
// per k-chunk 1 A load is reused across 4 WMMAs (1.25 KB/WMMA from cache).
// 64 v_wmma_f32_16x16x32_f16 per wave.
// ---------------------------------------------------------------------------
__global__ void gemm_wmma_kernel(const _Float16* __restrict__ Ap,
                                 const _Float16* __restrict__ Bp,
                                 float* __restrict__ scores, float scale) {
  int wave = threadIdx.x >> 5;
  int lane = threadIdx.x & 31;
  int bid = blockIdx.x;
  const int tSuper = NT / 32;           // 4 groups of 32 tTiles per block
  int tg    = bid % tSuper; bid /= tSuper;
  int sTile = bid % NT;
  int b     = bid / NT;
  int tTile0 = tg * 32 + wave * 4;

  const size_t kstride = (size_t)NT * 512;  // halves per k-chunk plane
  const _Float16* apb = Ap + (((size_t)b * KC * NT + sTile) * 32 + lane) * 16;
  const _Float16* bpb = Bp + (((size_t)b * KC * NT + tTile0) * 32 + lane) * 16;

  v8f acc0 = {}, acc1 = {}, acc2 = {}, acc3 = {};
  for (int kc = 0; kc < KC; ++kc) {
    const _Float16* bp = bpb + (size_t)kc * kstride;
    v16h a  = *(const v16h*)(apb + (size_t)kc * kstride);
    v16h b0 = *(const v16h*)(bp);
    v16h b1 = *(const v16h*)(bp + 512);
    v16h b2 = *(const v16h*)(bp + 1024);
    v16h b3 = *(const v16h*)(bp + 1536);
    acc0 = __builtin_amdgcn_wmma_f32_16x16x32_f16(false, a, false, b0, (short)0, acc0, false, false);
    acc1 = __builtin_amdgcn_wmma_f32_16x16x32_f16(false, a, false, b1, (short)0, acc1, false, false);
    acc2 = __builtin_amdgcn_wmma_f32_16x16x32_f16(false, a, false, b2, (short)0, acc2, false, false);
    acc3 = __builtin_amdgcn_wmma_f32_16x16x32_f16(false, a, false, b3, (short)0, acc3, false, false);
  }

  // C/D layout: VGPR r -> M = r + 8*(lane>=16); N = lane&15
  int t    = tTile0 * 16 + (lane & 15);
  int sRow = sTile * 16 + ((lane >> 4) << 3);
  float* out = scores + ((size_t)b * NN + sRow) * NN + t;
#pragma unroll
  for (int r = 0; r < 8; ++r) {
    float* o = out + (size_t)r * NN;
    o[0]  = acc0[r] * scale;
    o[16] = acc1[r] * scale;
    o[32] = acc2[r] * scale;
    o[48] = acc3[r] * scale;
  }
}

// ---------------------------------------------------------------------------
// Row softmax in place + fused per-row (max, first-argmax) of the normalized
// row (feeds the incremental matcher). One block per row.
// ---------------------------------------------------------------------------
__global__ void softmax_rows_kernel(float* __restrict__ scores,
                                    float* __restrict__ rowmaxG,
                                    int* __restrict__ rowargG) {
  int row = blockIdx.x;  // b*NN + s
  float* p = scores + (size_t)row * NN;
  int tid = threadIdx.x;
  __shared__ float redf[256];
  __shared__ int   redi[256];

  float m = -1e30f;
  for (int i = tid; i < NN; i += 256) m = fmaxf(m, p[i]);
  redf[tid] = m; __syncthreads();
  for (int s = 128; s > 0; s >>= 1) {
    if (tid < s) redf[tid] = fmaxf(redf[tid], redf[tid + s]);
    __syncthreads();
  }
  m = redf[0]; __syncthreads();

  float sum = 0.0f;
  for (int i = tid; i < NN; i += 256) {
    float e = __expf(p[i] - m);
    p[i] = e;
    sum += e;
  }
  redf[tid] = sum; __syncthreads();
  for (int s = 128; s > 0; s >>= 1) {
    if (tid < s) redf[tid] += redf[tid + s];
    __syncthreads();
  }
  float inv = 1.0f / redf[0];
  __syncthreads();

  // normalize + track (max, smallest-index argmax)
  float bm = -1.0f;
  int   bc = NN;
  for (int i = tid; i < NN; i += 256) {
    float v = p[i] * inv;
    p[i] = v;
    if (v > bm || (v == bm && i < bc)) { bm = v; bc = i; }
  }
  redf[tid] = bm; redi[tid] = bc; __syncthreads();
  for (int s = 128; s > 0; s >>= 1) {
    if (tid < s) {
      if (redf[tid + s] > redf[tid] ||
          (redf[tid + s] == redf[tid] && redi[tid + s] < redi[tid])) {
        redf[tid] = redf[tid + s]; redi[tid] = redi[tid + s];
      }
    }
    __syncthreads();
  }
  if (tid == 0) { rowmaxG[row] = redf[0]; rowargG[row] = redi[0]; }
}

// ---------------------------------------------------------------------------
// colsum[b][t] = sum_s scores[b][s][t]  (for src_corr.mean collapse)
// ---------------------------------------------------------------------------
__global__ void colsum_kernel(const float* __restrict__ scores,
                              float* __restrict__ colsum) {
  int b = blockIdx.y;
  int t = blockIdx.x * 256 + threadIdx.x;
  const float* p = scores + (size_t)b * NN * NN + t;
  float s = 0.0f;
  for (int i = 0; i < NN; ++i) s += p[(size_t)i * NN];
  colsum[b * NN + t] = s;
}

// ---------------------------------------------------------------------------
// Incremental greedy matcher. Per batch (one block, 1024 threads):
// keep per-row (max, first-argmax) in LDS; each of the 15 steps does a
// 2048-wide argmax over row maxima (value desc, row asc => first row-major
// max), suppresses the row+column, then rescans only rows whose cached
// argmax column was suppressed (one wave per rescan row).
// ---------------------------------------------------------------------------
__global__ void match_kernel(const float* __restrict__ scores,
                             const float* __restrict__ rowmaxG,
                             const int* __restrict__ rowargG,
                             int* __restrict__ rowsOut,
                             int* __restrict__ colsOut) {
  int b = blockIdx.x;
  int tid = threadIdx.x;
  const int T = 1024;
  const float* P = scores + (size_t)b * NN * NN;

  __shared__ float rmax[NN];
  __shared__ int   rarg[NN];
  __shared__ unsigned colUsed[NN / 32];
  __shared__ float bv[T];
  __shared__ int   bi[T];
  __shared__ int   listCnt;
  __shared__ int   list[NN];

  for (int i = tid; i < NN; i += T) {
    rmax[i] = rowmaxG[b * NN + i];
    rarg[i] = rowargG[b * NN + i];
  }
  for (int i = tid; i < NN / 32; i += T) colUsed[i] = 0u;
  __syncthreads();

  for (int step = 0; step < SS; ++step) {
    // global argmax over active rows (value desc, row asc)
    float best = -1.0f;
    int brow = NN;
    for (int i = tid; i < NN; i += T) {
      float v = rmax[i];
      if (v > best || (v == best && i < brow)) { best = v; brow = i; }
    }
    bv[tid] = best; bi[tid] = brow;
    __syncthreads();
    for (int s = T / 2; s > 0; s >>= 1) {
      if (tid < s) {
        if (bv[tid + s] > bv[tid] ||
            (bv[tid + s] == bv[tid] && bi[tid + s] < bi[tid])) {
          bv[tid] = bv[tid + s]; bi[tid] = bi[tid + s];
        }
      }
      __syncthreads();
    }
    int r = bi[0];
    int c = rarg[r];
    if (tid == 0) {
      rowsOut[b * SS + step] = r;
      colsOut[b * SS + step] = c;
      rmax[r] = -1.0f;                      // suppress row
      colUsed[c >> 5] |= 1u << (c & 31);    // suppress column
      listCnt = 0;
    }
    __syncthreads();

    // rows whose cached argmax column was just suppressed need a rescan
    for (int i = tid; i < NN; i += T) {
      if (rmax[i] >= 0.0f && rarg[i] == c) {
        int pos = atomicAdd(&listCnt, 1);
        list[pos] = i;
      }
    }
    __syncthreads();

    int nres = listCnt;
    int wid = tid >> 5, lane = tid & 31;
    for (int l = wid; l < nres; l += 32) {
      int row = list[l];
      const float* pr = P + (size_t)row * NN;
      float mv = -1.0f;
      int mc = NN;
      for (int cc = lane; cc < NN; cc += 32) {
        if ((colUsed[cc >> 5] >> (cc & 31)) & 1u) continue;
        float v = pr[cc];
        if (v > mv || (v == mv && cc < mc)) { mv = v; mc = cc; }
      }
      for (int off = 16; off > 0; off >>= 1) {
        float ov = __shfl_down(mv, off);
        int   oc = __shfl_down(mc, off);
        if (ov > mv || (ov == mv && oc < mc)) { mv = ov; mc = oc; }
      }
      if (lane == 0) { rmax[row] = mv; rarg[row] = mc; }
    }
    __syncthreads();
  }
}

// ---------------------------------------------------------------------------
// 3x3 Kabsch rotation: H = U S V^T via Jacobi on H^T H; r = V U^T with
// det<0 fix r = V diag(1,1,-1) U^T  (matches reference's reflect handling).
// ---------------------------------------------------------------------------
__device__ void kabsch3(const float H[3][3], float R[3][3]) {
  float A[3][3], V[3][3] = {{1, 0, 0}, {0, 1, 0}, {0, 0, 1}};
  for (int i = 0; i < 3; ++i)
    for (int j = 0; j < 3; ++j) {
      float s = 0.0f;
      for (int k = 0; k < 3; ++k) s += H[k][i] * H[k][j];
      A[i][j] = s;
    }
  const int pq[3][2] = {{0, 1}, {0, 2}, {1, 2}};
  for (int sweep = 0; sweep < 12; ++sweep)
    for (int r = 0; r < 3; ++r) {
      int p = pq[r][0], q = pq[r][1];
      float apq = A[p][q];
      if (fabsf(apq) < 1e-20f) continue;
      float th = (A[q][q] - A[p][p]) / (2.0f * apq);
      float t = copysignf(1.0f, th) / (fabsf(th) + sqrtf(1.0f + th * th));
      float c = 1.0f / sqrtf(1.0f + t * t), s = t * c;
      for (int k = 0; k < 3; ++k) {
        float akp = A[k][p], akq = A[k][q];
        A[k][p] = c * akp - s * akq; A[k][q] = s * akp + c * akq;
      }
      for (int k = 0; k < 3; ++k) {
        float apk = A[p][k], aqk = A[q][k];
        A[p][k] = c * apk - s * aqk; A[q][k] = s * apk + c * aqk;
      }
      for (int k = 0; k < 3; ++k) {
        float vkp = V[k][p], vkq = V[k][q];
        V[k][p] = c * vkp - s * vkq; V[k][q] = s * vkp + c * vkq;
      }
    }
  float lam[3] = {A[0][0], A[1][1], A[2][2]};
  int ord[3] = {0, 1, 2};
  for (int i = 0; i < 2; ++i)
    for (int j = i + 1; j < 3; ++j)
      if (lam[ord[j]] > lam[ord[i]]) { int t = ord[i]; ord[i] = ord[j]; ord[j] = t; }
  float Vs[3][3], U[3][3], sig[3];
  for (int k = 0; k < 3; ++k) {
    sig[k] = sqrtf(fmaxf(lam[ord[k]], 0.0f));
    for (int i = 0; i < 3; ++i) Vs[i][k] = V[i][ord[k]];
  }
  for (int k = 0; k < 3; ++k) {
    float uv[3];
    for (int i = 0; i < 3; ++i) {
      float s = 0.0f;
      for (int j = 0; j < 3; ++j) s += H[i][j] * Vs[j][k];
      uv[i] = s;
    }
    if (sig[k] > 1e-9f) {
      float inv = 1.0f / sig[k];
      for (int i = 0; i < 3; ++i) U[i][k] = uv[i] * inv;
    } else if (k == 2) {  // degenerate smallest: orthonormal completion
      U[0][2] = U[1][0] * U[2][1] - U[2][0] * U[1][1];
      U[1][2] = U[2][0] * U[0][1] - U[0][0] * U[2][1];
      U[2][2] = U[0][0] * U[1][1] - U[1][0] * U[0][1];
    } else {
      for (int i = 0; i < 3; ++i) U[i][k] = (i == k) ? 1.0f : 0.0f;
    }
  }
  float r0[3][3];
  for (int i = 0; i < 3; ++i)
    for (int j = 0; j < 3; ++j) {
      float s = 0.0f;
      for (int k = 0; k < 3; ++k) s += Vs[i][k] * U[j][k];
      r0[i][j] = s;
    }
  float det = r0[0][0] * (r0[1][1] * r0[2][2] - r0[1][2] * r0[2][1]) -
              r0[0][1] * (r0[1][0] * r0[2][2] - r0[1][2] * r0[2][0]) +
              r0[0][2] * (r0[1][0] * r0[2][1] - r0[1][1] * r0[2][0]);
  if (det < 0.0f) {
    const float dk[3] = {1.0f, 1.0f, -1.0f};
    for (int i = 0; i < 3; ++i)
      for (int j = 0; j < 3; ++j) {
        float s = 0.0f;
        for (int k = 0; k < 3; ++k) s += Vs[i][k] * dk[k] * U[j][k];
        R[i][j] = s;
      }
  } else {
    for (int i = 0; i < 3; ++i)
      for (int j = 0; j < 3; ++j) R[i][j] = r0[i][j];
  }
}

// ---------------------------------------------------------------------------
// Finalize: per batch compute mean(src), tgt@colsum/N, H from matched points,
// Kabsch R, t = -R*mean(src) + corr_mean. One block per batch.
// src_pts/tgt_pts are the 'bnc' inputs: [B][N][3].
// Output: R flattened [B*9] then t [B*3].
// ---------------------------------------------------------------------------
__global__ void finalize_kernel(const float* __restrict__ src_pts,
                                const float* __restrict__ tgt_pts,
                                const float* __restrict__ colsum,
                                const int* __restrict__ rowsIn,
                                const int* __restrict__ colsIn,
                                float* __restrict__ out) {
  int b = blockIdx.x;
  int tid = threadIdx.x;
  const float* sp = src_pts + (size_t)b * NN * 3;
  const float* tp = tgt_pts + (size_t)b * NN * 3;
  const float* cs = colsum + b * NN;

  __shared__ float sred[256][6];
  float acc[6] = {0, 0, 0, 0, 0, 0};
  for (int i = tid; i < NN; i += 256) {
    float w = cs[i];
    for (int j = 0; j < 3; ++j) {
      acc[j]     += sp[i * 3 + j];
      acc[3 + j] += tp[i * 3 + j] * w;
    }
  }
  for (int j = 0; j < 6; ++j) sred[tid][j] = acc[j];
  __syncthreads();
  for (int s = 128; s > 0; s >>= 1) {
    if (tid < s)
      for (int j = 0; j < 6; ++j) sred[tid][j] += sred[tid + s][j];
    __syncthreads();
  }

  if (tid == 0) {
    float meanS[3], corrM[3];
    for (int j = 0; j < 3; ++j) {
      meanS[j] = sred[0][j] * (1.0f / NN);
      corrM[j] = sred[0][3 + j] * (1.0f / NN);
    }
    float ps[3][SS], pt[3][SS], cS[3] = {0, 0, 0}, cT[3] = {0, 0, 0};
    for (int i = 0; i < SS; ++i) {
      int r = rowsIn[b * SS + i], c = colsIn[b * SS + i];
      for (int j = 0; j < 3; ++j) {
        ps[j][i] = sp[r * 3 + j];
        pt[j][i] = tp[c * 3 + j];
        cS[j] += ps[j][i];
        cT[j] += pt[j][i];
      }
    }
    for (int j = 0; j < 3; ++j) { cS[j] *= (1.0f / SS); cT[j] *= (1.0f / SS); }
    float H[3][3];
    for (int i = 0; i < 3; ++i)
      for (int j = 0; j < 3; ++j) {
        float s = 0.0f;
        for (int k = 0; k < SS; ++k)
          s += (ps[i][k] - cS[i]) * (pt[j][k] - cT[j]);
        H[i][j] = s;
      }
    float R[3][3];
    kabsch3(H, R);
    for (int i = 0; i < 3; ++i)
      for (int j = 0; j < 3; ++j) out[b * 9 + i * 3 + j] = R[i][j];
    float* tOut = out + BB * 9;
    for (int i = 0; i < 3; ++i) {
      float s = 0.0f;
      for (int j = 0; j < 3; ++j) s += R[i][j] * meanS[j];
      tOut[b * 3 + i] = -s + corrM[i];
    }
  }
}

// ---------------------------------------------------------------------------
extern "C" void kernel_launch(void* const* d_in, const int* in_sizes, int n_in,
                              void* d_out, int out_size, void* d_ws, size_t ws_size,
                              hipStream_t stream) {
  const float* src_emb = (const float*)d_in[0];  // [16,512,2048]
  const float* tgt_emb = (const float*)d_in[1];  // [16,512,2048]
  const float* src_pts = (const float*)d_in[2];  // [16,2048,3]
  const float* tgt_pts = (const float*)d_in[3];  // [16,2048,3]
  float* out = (float*)d_out;                    // 144 (R) + 48 (t)

  // workspace layout
  const size_t packElems = (size_t)BB * KC * NT * 32 * 16;  // halves
  _Float16* Ap = (_Float16*)d_ws;
  _Float16* Bp = Ap + packElems;
  float* scores  = (float*)(Bp + packElems);                // [16,2048,2048]
  float* colsum  = scores + (size_t)BB * NN * NN;           // [16,2048]
  float* rowmaxG = colsum + (size_t)BB * NN;                // [16,2048]
  int*   rowargG = (int*)(rowmaxG + (size_t)BB * NN);       // [16,2048]
  int*   rowsBuf = rowargG + BB * NN;                       // [16,15]
  int*   colsBuf = rowsBuf + BB * SS;

  const int nFrags = BB * KC * NT;  // 32768
  pack_frags_kernel<<<nFrags / 8, 256, 0, stream>>>(src_emb, Ap);
  pack_frags_kernel<<<nFrags / 8, 256, 0, stream>>>(tgt_emb, Bp);

  const float scale = 1.0f / sqrtf((float)DD);
  gemm_wmma_kernel<<<BB * NT * (NT / 32), 256, 0, stream>>>(Ap, Bp, scores, scale);

  softmax_rows_kernel<<<BB * NN, 256, 0, stream>>>(scores, rowmaxG, rowargG);

  colsum_kernel<<<dim3(NN / 256, BB), 256, 0, stream>>>(scores, colsum);

  match_kernel<<<BB, 1024, 0, stream>>>(scores, rowmaxG, rowargG,
                                        rowsBuf, colsBuf);

  finalize_kernel<<<BB, 256, 0, stream>>>(src_pts, tgt_pts, colsum,
                                          rowsBuf, colsBuf, out);
}
